// CausalSelfAttention_66760971649744
// MI455X (gfx1250) — matrix operand order repaired and checked by reading, not compiled, not run on hardware
//
#include <hip/hip_runtime.h>
#include <stdint.h>

// ---------------------------------------------------------------------------
// CDNA5 (gfx1250) causal local-window self-attention, f16 WMMA pipeline.
//   Stage 1: x -> f16 ; W_attn/W_proj -> f16 transposed [N][K]
//   Stage 2: QKV GEMM via WMMA, tiles fed by Tensor Data Mover (TDM),
//            double-buffered on TENSORcnt. Epilogue scatters Q,K head-major
//            and V transposed [B,H,D,T].
//   Stage 3: banded flash attention (window 256) with WMMA QK^T / PV,
//            Q tile staged via global_load_async_to_lds_b128 (ASYNCcnt).
//   Stage 4: projection GEMM (same TDM+WMMA core), fp32 epilogue.
// ---------------------------------------------------------------------------

typedef __attribute__((ext_vector_type(16))) _Float16 v16h;
typedef __attribute__((ext_vector_type(8)))  _Float16 v8h;
typedef __attribute__((ext_vector_type(4)))  _Float16 v4h;
typedef __attribute__((ext_vector_type(8)))  float    v8f;
typedef __attribute__((ext_vector_type(4)))  float    v4f;
typedef __attribute__((ext_vector_type(4)))  unsigned int v4u;
typedef __attribute__((ext_vector_type(8)))  int      v8i;
typedef __attribute__((ext_vector_type(4)))  int      v4i;

__device__ __forceinline__ v16h cat16(v8h lo, v8h hi) {
  v16h r;
#pragma unroll
  for (int i = 0; i < 8; ++i) { r[i] = lo[i]; r[i + 8] = hi[i]; }
  return r;
}

__device__ __forceinline__ v8f wmma_f16(v16h a, v16h b, v8f c) {
  return __builtin_amdgcn_wmma_f32_16x16x32_f16(
      false, a, false, b, (short)0, c, false, false);
}

// ---- Tensor Data Mover: 2D f16 tile, global -> LDS (D# per ISA ch.8) ------
__device__ __forceinline__ void tdm_load_2d_f16(
    unsigned lds_addr, const void* gptr,
    unsigned tile_d0 /*halves along k*/, unsigned tile_d1 /*rows*/,
    unsigned long long stride0 /*halves between rows*/) {
  const unsigned long long ga = (unsigned long long)(uintptr_t)gptr;
  v4u g0;
  g0[0] = 1u;                                    // count=1, user descriptor
  g0[1] = lds_addr;                              // LDS byte address
  g0[2] = (unsigned)(ga & 0xffffffffu);          // global_addr[31:0]
  g0[3] = (unsigned)((ga >> 32) & 0x01ffffffu)   // global_addr[56:32]
          | (2u << 30);                          // type = 2 ("image")
  const unsigned td0 = tile_d0, td1 = tile_d1;   // tensor dims == tile dims
  v8i g1;
  g1[0] = (int)(1u << 16);                       // data_size = 1 (2 bytes)
  g1[1] = (int)((td0 & 0xffffu) << 16);          // tensor_dim0[15:0]
  g1[2] = (int)((td0 >> 16) | ((td1 & 0xffffu) << 16));
  g1[3] = (int)((td1 >> 16) | (tile_d0 << 16));  // tile_dim0
  g1[4] = (int)(tile_d1);                        // tile_dim1 (tile_dim2 = 0)
  g1[5] = (int)(unsigned)(stride0 & 0xffffffffu);
  g1[6] = (int)(unsigned)((stride0 >> 32) & 0xffffu);  // stride1 = 0
  g1[7] = 0;
  const v4i gz4 = {0, 0, 0, 0};                  // 2D: groups 2/3 unused
  const v8i gz8 = {0, 0, 0, 0, 0, 0, 0, 0};
  __builtin_amdgcn_tensor_load_to_lds(g0, g1, gz4, gz4, gz8, 0);
}

// ---- async global -> LDS (16B per lane), tracked on ASYNCcnt --------------
__device__ __forceinline__ void async_load_b128(const void* gptr, void* lptr) {
  const unsigned lds = (unsigned)(uintptr_t)lptr;
  const unsigned long long ga = (unsigned long long)(uintptr_t)gptr;
  asm volatile("global_load_async_to_lds_b128 %0, %1, off"
               :: "v"(lds), "v"(ga) : "memory");
}
__device__ __forceinline__ void wait_async0() {
  asm volatile("s_wait_asynccnt 0" ::: "memory");
}
__device__ __forceinline__ void lds_fence() {
  asm volatile("s_wait_dscnt 0" ::: "memory");
}

// ---------------------------------------------------------------------------
// fp32 -> f16 linear conversion (x)
// ---------------------------------------------------------------------------
__global__ __launch_bounds__(256) void cvt_f32_f16(
    const float* __restrict__ src, _Float16* __restrict__ dst, int n4) {
  int i = blockIdx.x * 256 + threadIdx.x;
  if (i < n4) {
    v4f f = ((const v4f*)src)[i];
    ((v4h*)dst)[i] = __builtin_convertvector(f, v4h);
  }
}

// ---------------------------------------------------------------------------
// fp32 [K][N] -> f16 [N][K] tiled transpose (weights), 32x32 tiles
// ---------------------------------------------------------------------------
__global__ __launch_bounds__(256) void transpose_cvt(
    const float* __restrict__ src, _Float16* __restrict__ dst, int K, int N) {
  __shared__ _Float16 tile[32][33];
  const int k0 = blockIdx.y * 32, n0 = blockIdx.x * 32;
  const int tx = threadIdx.x & 31, ty = threadIdx.x >> 5;  // 32 x 8
#pragma unroll
  for (int j = 0; j < 32; j += 8)
    tile[ty + j][tx] = (_Float16)src[(size_t)(k0 + ty + j) * N + n0 + tx];
  __syncthreads();
#pragma unroll
  for (int j = 0; j < 32; j += 8)
    dst[(size_t)(n0 + ty + j) * K + k0 + tx] = tile[tx][ty + j];
}

// ---------------------------------------------------------------------------
// TDM-fed WMMA GEMM: C[M,N] = A[M,K] (f16 rm) * Bt[N,K] (f16, pre-transposed)
// BM=128, BN=128, BK=32; 8 waves, each 64x32 (4x2 WMMA accumulators).
// Double-buffered LDS, tiles moved by TENSOR_LOAD_TO_LDS from wave 0.
// MODE 0: bias + scatter Q,K head-major / V^T (f16).  MODE 1: bias + fp32.
// ---------------------------------------------------------------------------
#define BM 128
#define BN 128
#define BK 32

template <int MODE>
__global__ __launch_bounds__(256) void gemm_wmma_tdm(
    const _Float16* __restrict__ A, const _Float16* __restrict__ Bt,
    const float* __restrict__ bias, int M, int N, int K,
    _Float16* __restrict__ Qh, _Float16* __restrict__ Kh,
    _Float16* __restrict__ Vt, float* __restrict__ Cout) {
  __shared__ __align__(16) _Float16 As[2][BM * BK];   // [m][k]
  __shared__ __align__(16) _Float16 Bs[2][BN * BK];   // [n][k]

  const int tid  = threadIdx.x;
  const int lane = tid & 31;
  const int wave = tid >> 5;
  const int wy   = wave >> 2;   // 0..1 (M)
  const int wx   = wave & 3;    // 0..3 (N)
  const int g    = lane >> 4;
  const int ln   = lane & 15;
  const int kbA  = g ? 8 : 0;   // A/B fragment K base per ISA VGPR layout

  const int bm0 = blockIdx.y * BM;
  const int bn0 = blockIdx.x * BN;
  const int nIter = K / BK;

  const unsigned ldsA[2] = {(unsigned)(uintptr_t)&As[0][0],
                            (unsigned)(uintptr_t)&As[1][0]};
  const unsigned ldsB[2] = {(unsigned)(uintptr_t)&Bs[0][0],
                            (unsigned)(uintptr_t)&Bs[1][0]};

  v8f zero8 = {};
  v8f acc[4][2];
#pragma unroll
  for (int mi = 0; mi < 4; ++mi)
#pragma unroll
    for (int nj = 0; nj < 2; ++nj) acc[mi][nj] = zero8;

  if (wave == 0) {   // prologue: fill buffer 0
    tdm_load_2d_f16(ldsA[0], A  + (size_t)bm0 * K, BK, BM, (unsigned long long)K);
    tdm_load_2d_f16(ldsB[0], Bt + (size_t)bn0 * K, BK, BN, (unsigned long long)K);
  }

  for (int i = 0; i < nIter; ++i) {
    const int buf = i & 1;
    if (wave == 0) {
      if (i + 1 < nIter) {   // prefetch next tile into other buffer
        const int k1 = (i + 1) * BK;
        tdm_load_2d_f16(ldsA[buf ^ 1], A  + (size_t)bm0 * K + k1, BK, BM,
                        (unsigned long long)K);
        tdm_load_2d_f16(ldsB[buf ^ 1], Bt + (size_t)bn0 * K + k1, BK, BN,
                        (unsigned long long)K);
        __builtin_amdgcn_s_wait_tensorcnt(2);  // current 2 done, next 2 fly
      } else {
        __builtin_amdgcn_s_wait_tensorcnt(0);
      }
    }
    __syncthreads();

    v16h af[4], bf[2];
#pragma unroll
    for (int mi = 0; mi < 4; ++mi) {
      const _Float16* p = &As[buf][(wy * 64 + mi * 16 + ln) * BK + 16 * g];
      af[mi] = cat16(*(const v8h*)p, *(const v8h*)(p + 8));
    }
#pragma unroll
    for (int nj = 0; nj < 2; ++nj) {
      const _Float16* p = &Bs[buf][(wx * 32 + nj * 16 + ln) * BK + 16 * g];
      bf[nj] = cat16(*(const v8h*)p, *(const v8h*)(p + 8));
    }
#pragma unroll
    for (int mi = 0; mi < 4; ++mi)
#pragma unroll
      for (int nj = 0; nj < 2; ++nj)
        acc[mi][nj] = wmma_f16(af[mi], bf[nj], acc[mi][nj]);
    __syncthreads();   // reads done before next TDM overwrites this buffer
  }

  // ---- epilogue: C element (m = r + 8*g, n = ln) per accumulator VGPR r
#pragma unroll
  for (int mi = 0; mi < 4; ++mi)
#pragma unroll
    for (int nj = 0; nj < 2; ++nj) {
      const int gn = bn0 + wx * 32 + nj * 16 + ln;
      const float bv = bias[gn];
#pragma unroll
      for (int r = 0; r < 8; ++r) {
        const int gm = bm0 + wy * 64 + mi * 16 + r + 8 * g;
        const float v = acc[mi][nj][r] + bv;
        if (MODE == 0) {
          const int which = gn >> 10;   // 0=Q 1=K 2=V  (C = 1024)
          const int c = gn & 1023;
          const int h = c >> 6, d = c & 63;
          const int b = gm >> 11, t = gm & 2047;   // T = 2048
          const size_t bh = (size_t)(b * 16 + h);
          if (which == 0)      Qh[(bh * 2048 + t) * 64 + d] = (_Float16)v;
          else if (which == 1) Kh[(bh * 2048 + t) * 64 + d] = (_Float16)v;
          else                 Vt[(bh * 64 + d) * 2048 + t] = (_Float16)v;
        } else {
          Cout[(size_t)gm * N + gn] = v;
        }
      }
    }
}

// ---------------------------------------------------------------------------
// Banded flash attention (CTX=256). One wave = one 16-row query tile.
// Q tile staged with async global->LDS; K^T and V^T B-operands read straight
// from global (contiguous 16-half runs); P transposed C->A layout via LDS.
// ---------------------------------------------------------------------------
__global__ __launch_bounds__(256) void attn_wmma(
    const _Float16* __restrict__ Qh, const _Float16* __restrict__ Kh,
    const _Float16* __restrict__ Vt, _Float16* __restrict__ Aout) {
  constexpr int T = 2048, D = 64, CTX = 256, CC = 1024;
  __shared__ __align__(16) _Float16 Qs[8 * 16 * 64];
  __shared__ __align__(16) _Float16 Ps[8 * 16 * 32];

  const int tid  = threadIdx.x;
  const int lane = tid & 31;
  const int wave = tid >> 5;
  const int g    = lane >> 4;
  const int ln   = lane & 15;
  const int kb   = g ? 8 : 0;

  const int gid = blockIdx.x * 8 + wave;
  const int bh  = gid >> 7;                 // 128 query tiles per (b,h)
  const int t0  = (gid & 127) * 16;
  const int b   = bh >> 4;
  const int h   = bh & 15;

  const _Float16* Qb = Qh + (size_t)bh * T * D;
  const _Float16* Kb = Kh + (size_t)bh * T * D;
  const _Float16* Vb = Vt + (size_t)bh * D * T;
  _Float16* Qw = &Qs[wave * 16 * 64];
  _Float16* Pw = &Ps[wave * 16 * 32];

  // ---- stage Q tile (16x64) via async copy: 4 x 16B per lane
  {
    const int r = ln, c0 = g * 32;
    const _Float16* src = Qb + (size_t)(t0 + r) * D + c0;
    _Float16* dst = Qw + r * 64 + c0;
    async_load_b128(src,      dst);
    async_load_b128(src + 8,  dst + 8);
    async_load_b128(src + 16, dst + 16);
    async_load_b128(src + 24, dst + 24);
  }
  wait_async0();

  v16h aq[2];
#pragma unroll
  for (int ks = 0; ks < 2; ++ks) {
    const _Float16* qp = Qw + ln * 64 + ks * 32 + 16 * g;
    aq[ks] = cat16(*(const v8h*)qp, *(const v8h*)(qp + 8));
  }

  v8f zero8 = {};
  v16h zero16 = {};
  v8f O[4];
#pragma unroll
  for (int i = 0; i < 4; ++i) O[i] = zero8;
  float mst[8], lst[8];
#pragma unroll
  for (int r = 0; r < 8; ++r) { mst[r] = -3.0e38f; lst[r] = 0.f; }

  int s_lo = t0 - (CTX - 1);
  if (s_lo < 0) s_lo = 0;
  s_lo &= ~31;
  const int s_end = t0 + 16;

  for (int s0 = s_lo; s0 < s_end; s0 += 32) {
    // ---- S = Q K^T (two 16x16 tiles over 32 keys)
    v8f S[2]; S[0] = zero8; S[1] = zero8;
#pragma unroll
    for (int tn = 0; tn < 2; ++tn) {
      const int s = s0 + tn * 16 + ln;
#pragma unroll
      for (int ks = 0; ks < 2; ++ks) {
        v16h bk = zero16;
        if (s < T) {
          const _Float16* kp = Kb + (size_t)s * D + ks * 32 + g * 16;
          bk = cat16(*(const v8h*)kp, *(const v8h*)(kp + 8));
        }
        S[tn] = wmma_f16(aq[ks], bk, S[tn]);
      }
    }

    // ---- online softmax; rows r+8g, reduced within 16-lane halves
    const float scale = 0.125f;   // 1/sqrt(64)
    float al[8];
#pragma unroll
    for (int r = 0; r < 8; ++r) {
      const int t = t0 + r + 8 * g;
      float mx = -3.0e38f;
#pragma unroll
      for (int tn = 0; tn < 2; ++tn) {
        const int s = s0 + tn * 16 + ln;
        float v = S[tn][r] * scale;
        const bool ok = (s <= t) && ((t - s) < CTX);
        v = ok ? v : -3.0e38f;
        S[tn][r] = v;
        mx = fmaxf(mx, v);
      }
      mx = fmaxf(mx, __shfl_xor(mx, 1));
      mx = fmaxf(mx, __shfl_xor(mx, 2));
      mx = fmaxf(mx, __shfl_xor(mx, 4));
      mx = fmaxf(mx, __shfl_xor(mx, 8));
      const float mnew = fmaxf(mst[r], mx);
      const float a = __expf(mst[r] - mnew);
      mst[r] = mnew;
      al[r] = a;
      float rs = 0.f;
#pragma unroll
      for (int tn = 0; tn < 2; ++tn) {
        const float v = S[tn][r];
        const float p = (v > -1.0e30f) ? __expf(v - mnew) : 0.f;
        rs += p;
        Pw[(r + 8 * g) * 32 + tn * 16 + ln] = (_Float16)p;  // C->A transpose
      }
      rs += __shfl_xor(rs, 1);
      rs += __shfl_xor(rs, 2);
      rs += __shfl_xor(rs, 4);
      rs += __shfl_xor(rs, 8);
      lst[r] = lst[r] * a + rs;
    }

#pragma unroll
    for (int dt = 0; dt < 4; ++dt)
#pragma unroll
      for (int r = 0; r < 8; ++r) O[dt][r] *= al[r];

    lds_fence();

    // ---- O += P V  (V^T rows feed B-operand contiguously)
    const _Float16* pp = Pw + ln * 32 + 16 * g;
    const v16h pa = cat16(*(const v8h*)pp, *(const v8h*)(pp + 8));
    const int sc = s0 + 16 * g;
#pragma unroll
    for (int dt = 0; dt < 4; ++dt) {
      v16h bv = zero16;
      if (sc < T) {
        const _Float16* vp = Vb + (size_t)(dt * 16 + ln) * T + sc;
        bv = cat16(*(const v8h*)vp, *(const v8h*)(vp + 8));
      }
      O[dt] = wmma_f16(pa, bv, O[dt]);
    }
    lds_fence();   // P reads retired before next iteration's P writes
  }

  // ---- normalize, write attention output f16 [B*T, C]
#pragma unroll
  for (int dt = 0; dt < 4; ++dt) {
    const int d = dt * 16 + ln;
#pragma unroll
    for (int r = 0; r < 8; ++r) {
      const int t = t0 + r + 8 * g;
      const float v = O[dt][r] / lst[r];
      Aout[((size_t)(b * T + t)) * CC + h * 64 + d] = (_Float16)v;
    }
  }
}

// ---------------------------------------------------------------------------
// Host-side launcher.  Workspace (f16, 256B aligned), ~92 MB total.
// ---------------------------------------------------------------------------
extern "C" void kernel_launch(void* const* d_in, const int* in_sizes, int n_in,
                              void* d_out, int out_size, void* d_ws,
                              size_t ws_size, hipStream_t stream) {
  (void)in_sizes; (void)n_in; (void)out_size; (void)ws_size;
  const float* x      = (const float*)d_in[0];
  const float* W_attn = (const float*)d_in[1];
  const float* b_attn = (const float*)d_in[2];
  const float* W_proj = (const float*)d_in[3];
  const float* b_proj = (const float*)d_in[4];
  float* out = (float*)d_out;

  const int B = 4, T = 2048, C = 1024, H = 16;
  const size_t nx  = (size_t)B * T * C;
  const size_t nwa = (size_t)C * 3 * C;
  const size_t nwp = (size_t)C * C;

  char* ws = (char*)d_ws;
  size_t off = 0;
  auto take = [&](size_t halves) -> _Float16* {
    _Float16* p = (_Float16*)(ws + off);
    off += (halves * sizeof(_Float16) + 255) & ~(size_t)255;
    return p;
  };
  _Float16* Xh   = take(nx);
  _Float16* WahT = take(nwa);   // W_attn^T  [3072][1024] f16
  _Float16* WphT = take(nwp);   // W_proj^T  [1024][1024] f16
  _Float16* Qh   = take(nx);
  _Float16* Kh   = take(nx);
  _Float16* Vt   = take(nx);
  _Float16* Ah   = take(nx);

  // 1) conversions: x linear; weights transposed to [N][K]
  cvt_f32_f16<<<(int)((nx / 4 + 255) / 256), 256, 0, stream>>>(x, Xh, (int)(nx / 4));
  {
    dim3 gt(3 * C / 32, C / 32);   // (96, 32)
    transpose_cvt<<<gt, 256, 0, stream>>>(W_attn, WahT, C, 3 * C);
  }
  {
    dim3 gt(C / 32, C / 32);       // (32, 32)
    transpose_cvt<<<gt, 256, 0, stream>>>(W_proj, WphT, C, C);
  }

  // 2) QKV GEMM + scatter epilogue
  {
    dim3 grid(3 * C / BN, B * T / BM);   // (24, 64)
    gemm_wmma_tdm<0><<<grid, 256, 0, stream>>>(Xh, WahT, b_attn, B * T, 3 * C,
                                               C, Qh, Kh, Vt, nullptr);
  }

  // 3) banded flash attention: 8192 query tiles, 8 waves / block
  attn_wmma<<<B * H * (T / 16) / 8, 256, 0, stream>>>(Qh, Kh, Vt, Ah);

  // 4) projection GEMM, fp32 out
  {
    dim3 grid(C / BN, B * T / BM);       // (8, 64)
    gemm_wmma_tdm<1><<<grid, 256, 0, stream>>>(Ah, WphT, b_proj, B * T, C, C,
                                               nullptr, nullptr, nullptr, out);
  }
}
